// GNNCoverageModel_12292196402001
// MI455X (gfx1250) — matrix operand order repaired and checked by reading, not compiled, not run on hardware
//
#include <hip/hip_runtime.h>

typedef __attribute__((ext_vector_type(16))) _Float16 v16h;
typedef __attribute__((ext_vector_type(8)))  float    v8f;

#define HID    128
#define INDIM  16
#define NGRAPH 64
#define OUTD   16   // 8 robots * 2

// ---------------- utility kernels ----------------

__global__ void gcn_zero_f32(float* __restrict__ p, int n) {
  int i = blockIdx.x * blockDim.x + threadIdx.x;
  if (i < n) p[i] = 0.0f;
}

__global__ void gcn_cvt_weights(const float* __restrict__ W1, const float* __restrict__ W2,
                                _Float16* __restrict__ W1h, _Float16* __restrict__ W2h) {
  int i = blockIdx.x * blockDim.x + threadIdx.x;
  if (i < INDIM * HID) W1h[i] = (_Float16)W1[i];
  if (i < HID * HID)   W2h[i] = (_Float16)W2[i];
}

__global__ void gcn_deg_count(const int* __restrict__ dst, float* __restrict__ deg, int E) {
  int e = blockIdx.x * blockDim.x + threadIdx.x;
  if (e < E) atomicAdd(&deg[dst[e]], 1.0f);
}

__global__ void gcn_dinv_fin(float* __restrict__ deg, int n) {
  int i = blockIdx.x * blockDim.x + threadIdx.x;
  if (i < n) deg[i] = rsqrtf(deg[i] + 1.0f);  // +1 for self-loop; always > 0
}

// ---------------- WMMA GEMM: h = x @ W1   (K = 16, zero-padded to 32) ----------------
// grid.x = n_nodes/16 (M tiles), 256 threads = 8 waves, wave w owns N-tile w.

__global__ void __launch_bounds__(256)
gcn_gemm1_wmma(const float* __restrict__ x, const _Float16* __restrict__ Wh,
               float* __restrict__ out) {
  const int mtile = blockIdx.x;
  const int lane  = threadIdx.x & 31;
  const int ntile = threadIdx.x >> 5;             // 0..7
  const int m     = mtile * 16 + (lane & 15);
  const int kbase = (lane < 16) ? 0 : 8;

  // A fragment: row m, halfs 0..7 = K kbase..kbase+7 ; halfs 8..15 (K=16..31) zero-pad.
  const float4* xr = (const float4*)(x + m * INDIM + kbase);
  float4 p0 = xr[0], p1 = xr[1];
  v16h a = {};
  a[0] = (_Float16)p0.x; a[1] = (_Float16)p0.y; a[2] = (_Float16)p0.z; a[3] = (_Float16)p0.w;
  a[4] = (_Float16)p1.x; a[5] = (_Float16)p1.y; a[6] = (_Float16)p1.z; a[7] = (_Float16)p1.w;

  // B fragment: column n = lane%16; halfs 0..7 = K kbase..kbase+7 ; halfs 8..15 (K=16..31) zero-pad.
  v16h b = {};
  for (int s = 0; s < 8; ++s) b[s] = Wh[(kbase + s) * HID + ntile * 16 + (lane & 15)];

  v8f c = {};
  c = __builtin_amdgcn_wmma_f32_16x16x32_f16(false, a, false, b, (short)0, c, false, false);

  float* orow = out + (mtile * 16 + ((lane < 16) ? 0 : 8)) * HID + ntile * 16 + (lane & 15);
#pragma unroll
  for (int v = 0; v < 8; ++v) orow[v * HID] = c[v];
}

// ---------------- WMMA GEMM: out = h @ W2   (K = 128 -> 4 chained WMMAs) ----------------

__global__ void __launch_bounds__(256)
gcn_gemm2_wmma(const float* __restrict__ hin, const _Float16* __restrict__ Wh,
               float* __restrict__ out) {
  const int mtile = blockIdx.x;
  const int lane  = threadIdx.x & 31;
  const int ntile = threadIdx.x >> 5;
  const int m     = mtile * 16 + (lane & 15);
  const int klo   = (lane < 16) ? 0 : 8;
  const int khi   = (lane < 16) ? 16 : 24;

  v8f c = {};
#pragma unroll
  for (int k0 = 0; k0 < HID; k0 += 32) {
    const float4* plo = (const float4*)(hin + m * HID + k0 + klo);
    const float4* phi = (const float4*)(hin + m * HID + k0 + khi);
    float4 q0 = plo[0], q1 = plo[1], q2 = phi[0], q3 = phi[1];
    v16h a;
    a[0]  = (_Float16)q0.x; a[1]  = (_Float16)q0.y; a[2]  = (_Float16)q0.z; a[3]  = (_Float16)q0.w;
    a[4]  = (_Float16)q1.x; a[5]  = (_Float16)q1.y; a[6]  = (_Float16)q1.z; a[7]  = (_Float16)q1.w;
    a[8]  = (_Float16)q2.x; a[9]  = (_Float16)q2.y; a[10] = (_Float16)q2.z; a[11] = (_Float16)q2.w;
    a[12] = (_Float16)q3.x; a[13] = (_Float16)q3.y; a[14] = (_Float16)q3.z; a[15] = (_Float16)q3.w;

    v16h b;  // column n = lane%16; halfs 0..7 = K klo.., halfs 8..15 = K khi..
    for (int s = 0; s < 8; ++s) {
      b[s]     = Wh[(k0 + klo + s) * HID + ntile * 16 + (lane & 15)];
      b[s + 8] = Wh[(k0 + khi + s) * HID + ntile * 16 + (lane & 15)];
    }
    c = __builtin_amdgcn_wmma_f32_16x16x32_f16(false, a, false, b, (short)0, c, false, false);
  }

  float* orow = out + (mtile * 16 + ((lane < 16) ? 0 : 8)) * HID + ntile * 16 + (lane & 15);
#pragma unroll
  for (int v = 0; v < 8; ++v) orow[v * HID] = c[v];
}

// ---------------- aggregation ----------------
// init: out[i] = dinv[i]^2 * h[i] + bias   (self-loop message + bias)
__global__ void gcn_agg_init(const float* __restrict__ h, const float* __restrict__ dinv,
                             const float* __restrict__ bias, float* __restrict__ out, int n) {
  int idx  = blockIdx.x * blockDim.x + threadIdx.x;
  int node = idx >> 5;                  // 32 lanes/node, float4 each = 128 feats
  int l    = idx & 31;
  if (node >= n) return;
  float di = dinv[node];
  float w  = di * di;
  float4 v  = *(const float4*)(h + node * HID + l * 4);
  float4 bb = *(const float4*)(bias + l * 4);
  float4 r  = {w * v.x + bb.x, w * v.y + bb.y, w * v.z + bb.z, w * v.w + bb.w};
  *(float4*)(out + node * HID + l * 4) = r;
}

// edges: out[dst] += dinv[src]*dinv[dst] * h[src]   (one wave per edge)
__global__ void gcn_agg_edges(const float* __restrict__ h, const float* __restrict__ dinv,
                              const int* __restrict__ src, const int* __restrict__ dst,
                              float* __restrict__ out, int E) {
  int idx = blockIdx.x * blockDim.x + threadIdx.x;
  int e   = idx >> 5;
  if (e >= E) return;
  int l = idx & 31;
  int s = src[e], d = dst[e];
  float w = dinv[s] * dinv[d];
  float4 v = *(const float4*)(h + s * HID + l * 4);
  float* o = out + d * HID + l * 4;
  atomicAdd(o + 0, w * v.x);
  atomicAdd(o + 1, w * v.y);
  atomicAdd(o + 2, w * v.z);
  atomicAdd(o + 3, w * v.w);
}

__global__ void gcn_relu4(float* __restrict__ p, int n4) {
  int i = blockIdx.x * blockDim.x + threadIdx.x;
  if (i >= n4) return;
  float4 v = *(float4*)(p + i * 4);
  v.x = fmaxf(v.x, 0.f); v.y = fmaxf(v.y, 0.f); v.z = fmaxf(v.z, 0.f); v.w = fmaxf(v.w, 0.f);
  *(float4*)(p + i * 4) = v;
}

// ---------------- pooling + FC ----------------

__global__ void gcn_pool(const float* __restrict__ h, const int* __restrict__ batch,
                         float* __restrict__ pooled, float* __restrict__ cnt, int n) {
  int idx  = blockIdx.x * blockDim.x + threadIdx.x;
  int node = idx >> 5;
  if (node >= n) return;
  int l = idx & 31;
  int g = batch[node];
  float4 v = *(const float4*)(h + node * HID + l * 4);
  float* p = pooled + g * HID + l * 4;
  atomicAdd(p + 0, v.x);
  atomicAdd(p + 1, v.y);
  atomicAdd(p + 2, v.z);
  atomicAdd(p + 3, v.w);
  if (l == 0) atomicAdd(cnt + g, 1.0f);
}

__global__ void gcn_fc(const float* __restrict__ pooled, const float* __restrict__ cnt,
                       const float* __restrict__ Wfc, const float* __restrict__ bfc,
                       float* __restrict__ out) {
  int idx = blockIdx.x * blockDim.x + threadIdx.x;   // 64*16 outputs
  if (idx >= NGRAPH * OUTD) return;
  int g = idx / OUTD, o = idx % OUTD;
  float inv = 1.0f / fmaxf(cnt[g], 1.0f);
  float acc = bfc[o];
  const float* pr = pooled + g * HID;
#pragma unroll 4
  for (int f = 0; f < HID; ++f) acc += pr[f] * inv * Wfc[f * OUTD + o];
  out[idx] = acc;
}

// ---------------- launch ----------------

extern "C" void kernel_launch(void* const* d_in, const int* in_sizes, int n_in,
                              void* d_out, int out_size, void* d_ws, size_t ws_size,
                              hipStream_t stream) {
  const float* x     = (const float*)d_in[0];
  const int*   eidx  = (const int*)d_in[1];
  const int*   batch = (const int*)d_in[2];
  const float* W1    = (const float*)d_in[3];
  const float* b1    = (const float*)d_in[4];
  const float* W2    = (const float*)d_in[5];
  const float* b2    = (const float*)d_in[6];
  const float* Wfc   = (const float*)d_in[7];
  const float* bfc   = (const float*)d_in[8];
  float* out = (float*)d_out;

  const int N = in_sizes[0] / INDIM;     // 50000
  const int E = in_sizes[1] / 2;         // 800000
  const int* src = eidx;
  const int* dst = eidx + E;

  // carve workspace (256B aligned slices)
  char* ws = (char*)d_ws;
  size_t off = 0;
  auto take = [&](size_t bytes) -> char* {
    char* p = ws + off;
    off = (off + bytes + 255) & ~(size_t)255;
    return p;
  };
  _Float16* W1h  = (_Float16*)take(sizeof(_Float16) * INDIM * HID);
  _Float16* W2h  = (_Float16*)take(sizeof(_Float16) * HID * HID);
  float* dinv    = (float*)take(sizeof(float) * N);
  float* buf0    = (float*)take(sizeof(float) * (size_t)N * HID);
  float* buf1    = (float*)take(sizeof(float) * (size_t)N * HID);
  float* pooled  = (float*)take(sizeof(float) * NGRAPH * HID);
  float* cnt     = (float*)take(sizeof(float) * NGRAPH);
  (void)ws_size;

  const int TB = 256;
  const int mtiles   = N / 16;                    // 3125
  const int nodeWarp = (N * 32 + TB - 1) / TB;    // 32 lanes per node
  const int edgeWarp = (E * 32 + TB - 1) / TB;
  const int elem4    = (N * HID / 4 + TB - 1) / TB;

  // weights -> fp16 ; zero deg/pooled/cnt
  gcn_cvt_weights<<<(HID * HID + TB - 1) / TB, TB, 0, stream>>>(W1, W2, W1h, W2h);
  gcn_zero_f32<<<(N + TB - 1) / TB, TB, 0, stream>>>(dinv, N);
  gcn_zero_f32<<<(NGRAPH * HID + TB - 1) / TB, TB, 0, stream>>>(pooled, NGRAPH * HID);
  gcn_zero_f32<<<1, TB, 0, stream>>>(cnt, NGRAPH);

  // degree -> dinv
  gcn_deg_count<<<(E + TB - 1) / TB, TB, 0, stream>>>(dst, dinv, E);
  gcn_dinv_fin<<<(N + TB - 1) / TB, TB, 0, stream>>>(dinv, N);

  // layer 1: h = relu(agg(x @ W1) + b1)
  gcn_gemm1_wmma<<<mtiles, TB, 0, stream>>>(x, W1h, buf0);
  gcn_agg_init<<<nodeWarp, TB, 0, stream>>>(buf0, dinv, b1, buf1, N);
  gcn_agg_edges<<<edgeWarp, TB, 0, stream>>>(buf0, dinv, src, dst, buf1, E);
  gcn_relu4<<<elem4, TB, 0, stream>>>(buf1, N * HID / 4);

  // layer 2: h = relu(agg(h @ W2) + b2)
  gcn_gemm2_wmma<<<mtiles, TB, 0, stream>>>(buf1, W2h, buf0);
  gcn_agg_init<<<nodeWarp, TB, 0, stream>>>(buf0, dinv, b2, buf1, N);
  gcn_agg_edges<<<edgeWarp, TB, 0, stream>>>(buf0, dinv, src, dst, buf1, E);
  gcn_relu4<<<elem4, TB, 0, stream>>>(buf1, N * HID / 4);

  // mean pool + FC
  gcn_pool<<<nodeWarp, TB, 0, stream>>>(buf1, batch, pooled, cnt, N);
  gcn_fc<<<(NGRAPH * OUTD + TB - 1) / TB, TB, 0, stream>>>(pooled, cnt, Wfc, bfc, out);
}
